// Model_5471788335740
// MI455X (gfx1250) — matrix operand order repaired and checked by reading, not compiled, not run on hardware
//
#include <hip/hip_runtime.h>
#include <math.h>

// ---------------------------------------------------------------------------
// Problem constants (from the reference)
// ---------------------------------------------------------------------------
#define N_NODES 50000     // N_M == N_J ; 50000 = 3125 * 16 (exact M-tiling)
#define N_EDGES 400000
#define IN_FEAT 16
#define LIN1    8
#define HEADS   3
#define HID     32
#define HC      96        // HEADS*HID
#define EDGE_DIM 5
#define NEG_SLOPE 0.2f

typedef __attribute__((ext_vector_type(16))) _Float16 v16h;
typedef __attribute__((ext_vector_type(8)))  float    v8f;

// ---------------------------------------------------------------------------
// WMMA GEMM:  Y[Nrows, NCOLS] = X[Nrows, K] @ W[K, NCOLS] + bias, optional tanh.
// K, NCOLS compile-time ->  all range checks fold to constants or value-selects
// (v_cndmask), loads are unconditional float4 (global_load_b128). One wave per
// 16x16 output tile, v_wmma_f32_16x16x32_f16, f32 accumulation.
//
// A 16x32 f16 layout (ISA 7.12.2): lane L -> M = L&15, kb = (L>>4)*8;
//   elem e<8 : K = k0+kb+e ; e>=8 : K = k0+kb+8+e
// B 32x16 f16 layout: lane L -> N = L&15, elem e -> same K as A (kb = (L>>4)*8).
// D f32 (8 VGPRs): VGPR r: lanes 0-15 -> (M=r, N=lane), lanes 16-31 -> (M=r+8).
//
// Requirements honored by call sites:
//  - Nrows % 16 == 0
//  - for K==8 the X buffer has >= 8 floats of readable tail (workspace buffers)
// ---------------------------------------------------------------------------
template<int K, int NCOLS, int TANH>
__global__ __launch_bounds__(128)
void gemm_wmma_kernel(const float* __restrict__ X,
                      const float* __restrict__ W,     // [K, NCOLS] row-major
                      const float* __restrict__ bias,  // [NCOLS]
                      float* __restrict__ Y,           // [Nrows, NCOLS]
                      int mtiles)
{
    constexpr int KSTEPS = (K + 31) / 32;
    constexpr int NTILES = (NCOLS + 15) / 16;

    int wave = (blockIdx.x * blockDim.x + threadIdx.x) >> 5;   // global wave id
    int lane = threadIdx.x & 31;
    if (wave >= mtiles * NTILES) return;                       // wave-uniform exit
    int mt = wave / NTILES;
    int nt = wave % NTILES;

    int am = mt * 16 + (lane & 15);      // A row for this lane (always < Nrows)
    int kb = (lane >> 4) * 8;            // A k-half base (0 or 8)
    const float* Arow = X + (long)am * K;

    v8f c = {};
    #pragma unroll
    for (int s = 0; s < KSTEPS; ++s) {
        const int k0 = s * 32;

        // ---------------- A fragment ----------------
        float af[16];
        if (K % 32 == 0) {
            // fully in-range: two contiguous 8-float chunks per lane
            const float* p = Arow + k0 + kb;
            float4 q0 = *(const float4*)(p);
            float4 q1 = *(const float4*)(p + 4);
            float4 q2 = *(const float4*)(p + 16);
            float4 q3 = *(const float4*)(p + 20);
            af[0]=q0.x; af[1]=q0.y; af[2]=q0.z; af[3]=q0.w;
            af[4]=q1.x; af[5]=q1.y; af[6]=q1.z; af[7]=q1.w;
            af[8]=q2.x; af[9]=q2.y; af[10]=q2.z; af[11]=q2.w;
            af[12]=q3.x; af[13]=q3.y; af[14]=q3.z; af[15]=q3.w;
        } else if (K == 16) {
            // elems 0..7 valid for both lane halves (kb+7 <= 15); 8..15 are K>=16 -> 0
            const float* p = Arow + kb;
            float4 q0 = *(const float4*)(p);
            float4 q1 = *(const float4*)(p + 4);
            af[0]=q0.x; af[1]=q0.y; af[2]=q0.z; af[3]=q0.w;
            af[4]=q1.x; af[5]=q1.y; af[6]=q1.z; af[7]=q1.w;
            #pragma unroll
            for (int e = 8; e < 16; ++e) af[e] = 0.0f;
        } else {
            // K == 8: kb==8 lanes read past the row (workspace tail, safe) and
            // are zeroed with a value select -- no EXEC branching.
            const float* p = Arow + kb;
            float4 q0 = *(const float4*)(p);
            float4 q1 = *(const float4*)(p + 4);
            bool ok = (kb == 0);
            af[0]= ok?q0.x:0.f; af[1]= ok?q0.y:0.f; af[2]= ok?q0.z:0.f; af[3]= ok?q0.w:0.f;
            af[4]= ok?q1.x:0.f; af[5]= ok?q1.y:0.f; af[6]= ok?q1.z:0.f; af[7]= ok?q1.w:0.f;
            #pragma unroll
            for (int e = 8; e < 16; ++e) af[e] = 0.0f;
        }
        v16h a;
        #pragma unroll
        for (int e = 0; e < 16; ++e) a[e] = (_Float16)af[e];

        // ---------------- B fragment ----------------
        float bf[16];
        {
            int bn = nt * 16 + (lane & 15);
            bool nok = (NCOLS % 16 == 0) || (bn < NCOLS);
            int bnc = nok ? bn : 0;
            #pragma unroll
            for (int e = 0; e < 16; ++e) {
                int kr = k0 + kb + (e < 8 ? e : e + 8);
                bool kok = (K % 32 == 0) || (kr < K);
                int krc = kok ? kr : 0;
                float v = W[(long)krc * NCOLS + bnc];
                bf[e] = (kok && nok) ? v : 0.0f;
            }
        }
        v16h b;
        #pragma unroll
        for (int e = 0; e < 16; ++e) b[e] = (_Float16)bf[e];

        // D = A*B + C   (neg_a, A, neg_b, B, c_mod, C, reuse_a, reuse_b)
        c = __builtin_amdgcn_wmma_f32_16x16x32_f16(false, a, false, b,
                                                   (short)0, c, false, false);
    }

    // ---------------- store D (+bias, optional tanh) ----------------
    int rn  = nt * 16 + (lane & 15);
    int rmo = (lane >> 4) * 8;
    if (NCOLS % 16 == 0 || rn < NCOLS) {
        float bv = bias[rn];
        #pragma unroll
        for (int r = 0; r < 8; ++r) {
            int rm = mt * 16 + rmo + r;          // always < Nrows (Nrows%16==0)
            float v = c[r] + bv;
            if (TANH) v = tanhf(v);
            Y[(long)rm * NCOLS + rn] = v;
        }
    }
}

// ---------------------------------------------------------------------------
// Utility kernels
// ---------------------------------------------------------------------------
__global__ void fill_kernel(float* __restrict__ p, float v, int n) {
    int i = blockIdx.x * blockDim.x + threadIdx.x;
    if (i < n) p[i] = v;
}

// out[n, 0:96] = bias[0:96]  (isolated dst nodes correctly end up = bias)
__global__ void init_bias_kernel(float* __restrict__ out, const float* __restrict__ bias, int N) {
    int i = blockIdx.x * blockDim.x + threadIdx.x;
    if (i < N * HC) out[i] = bias[i % HC];
}

__device__ inline void atomicMaxFloat(float* addr, float val) {
    int old = __float_as_int(*addr);
    while (__int_as_float(old) < val) {
        int prev = atomicCAS((int*)addr, old, __float_as_int(val));
        if (prev == old) break;
        old = prev;
    }
}

// ---------------------------------------------------------------------------
// Edge pass 1: score[e,h] = sum_c att[h,c]*leakyrelu(xs[src]+xd[dst]+eattr@We)
// plus atomic running max per (dst, head). One wave per edge; lane = channel%32.
// ---------------------------------------------------------------------------
__global__ void edge_score_kernel(const int* __restrict__ src, const int* __restrict__ dst,
                                  const float* __restrict__ eattr,   // [E,5]
                                  const float* __restrict__ We,      // [5,96]
                                  const float* __restrict__ att,     // [3,32]
                                  const float* __restrict__ xs,      // [Nsrc,96]
                                  const float* __restrict__ xd,      // [Ndst,96]
                                  float* __restrict__ score,         // [E,3]
                                  float* __restrict__ smax,          // [Ndst,3], init -inf
                                  int E)
{
    int e = (blockIdx.x * blockDim.x + threadIdx.x) >> 5;
    int lane = threadIdx.x & 31;
    if (e >= E) return;
    int s = src[e], d = dst[e];
    float ea[EDGE_DIM];
    #pragma unroll
    for (int i = 0; i < EDGE_DIM; ++i) ea[i] = eattr[e * EDGE_DIM + i];

    float sc[HEADS];
    #pragma unroll
    for (int h = 0; h < HEADS; ++h) {
        int c = h * HID + lane;
        float ep = 0.0f;
        #pragma unroll
        for (int i = 0; i < EDGE_DIM; ++i) ep += ea[i] * We[i * HC + c];
        float z = xs[s * HC + c] + xd[d * HC + c] + ep;
        z = (z > 0.0f) ? z : NEG_SLOPE * z;
        float v = z * att[h * HID + lane];
        #pragma unroll
        for (int off = 16; off > 0; off >>= 1) v += __shfl_xor(v, off, 32);
        sc[h] = v;
    }
    if (lane == 0) {
        #pragma unroll
        for (int h = 0; h < HEADS; ++h) {
            score[e * HEADS + h] = sc[h];
            atomicMaxFloat(&smax[d * HEADS + h], sc[h]);
        }
    }
}

// Edge pass 2: ex = exp(score - smax[dst]); denom[dst] += ex. One thread per edge.
__global__ void edge_exp_kernel(const int* __restrict__ dst,
                                float* __restrict__ score,          // in: score, out: ex
                                const float* __restrict__ smax,
                                float* __restrict__ denom, int E)
{
    int e = blockIdx.x * blockDim.x + threadIdx.x;
    if (e >= E) return;
    int d = dst[e];
    #pragma unroll
    for (int h = 0; h < HEADS; ++h) {
        float ex = expf(score[e * HEADS + h] - smax[d * HEADS + h]);
        score[e * HEADS + h] = ex;
        atomicAdd(&denom[d * HEADS + h], ex);
    }
}

// Edge pass 3: out[dst, c] += xs[src, c] * ex/(denom+1e-16). One wave per edge.
__global__ void edge_scatter_kernel(const int* __restrict__ src, const int* __restrict__ dst,
                                    const float* __restrict__ xs,
                                    const float* __restrict__ ex,     // [E,3]
                                    const float* __restrict__ denom,  // [Ndst,3]
                                    float* __restrict__ out, int E)
{
    int e = (blockIdx.x * blockDim.x + threadIdx.x) >> 5;
    int lane = threadIdx.x & 31;
    if (e >= E) return;
    int s = src[e], d = dst[e];
    #pragma unroll
    for (int h = 0; h < HEADS; ++h) {
        float alpha = ex[e * HEADS + h] / (denom[d * HEADS + h] + 1e-16f);
        int c = h * HID + lane;
        atomicAdd(&out[d * HC + c], xs[s * HC + c] * alpha);
    }
}

// p[n] = dot(tanh(H[n, 0:96]), w[0:96]).  One wave per node.
__global__ void node_dot_kernel(const float* __restrict__ H, const float* __restrict__ w,
                                float* __restrict__ p, int N)
{
    int n = (blockIdx.x * blockDim.x + threadIdx.x) >> 5;
    int lane = threadIdx.x & 31;
    if (n >= N) return;
    float v = 0.0f;
    #pragma unroll
    for (int h = 0; h < HEADS; ++h) {
        int c = h * HID + lane;
        v += tanhf(H[n * HC + c]) * w[c];
    }
    #pragma unroll
    for (int off = 16; off > 0; off >>= 1) v += __shfl_xor(v, off, 32);
    if (lane == 0) p[n] = v;
}

// out[e] = pm[src] + pj[dst] + eattr[e,:]·W3[96:101] + b3
__global__ void edge_out_kernel(const int* __restrict__ src, const int* __restrict__ dst,
                                const float* __restrict__ eattr,
                                const float* __restrict__ W3,   // 197 floats
                                const float* __restrict__ b3,
                                const float* __restrict__ pm, const float* __restrict__ pj,
                                float* __restrict__ out, int E)
{
    int e = blockIdx.x * blockDim.x + threadIdx.x;
    if (e >= E) return;
    float acc = pm[src[e]] + pj[dst[e]] + b3[0];
    #pragma unroll
    for (int i = 0; i < EDGE_DIM; ++i) acc += eattr[e * EDGE_DIM + i] * W3[HC + i];
    out[e] = acc;
}

// ---------------------------------------------------------------------------
// Host-side helpers
// ---------------------------------------------------------------------------
template<int K, int NCOLS, int TANH>
static inline void launch_gemm(const float* X, const float* W, const float* bias,
                               float* Y, int Nrows, hipStream_t s)
{
    constexpr int NTILES = (NCOLS + 15) / 16;
    int mtiles = Nrows / 16;                  // Nrows % 16 == 0 by construction
    int waves  = mtiles * NTILES;
    int blocks = (waves + 3) / 4;             // 128 threads = 4 waves per block
    gemm_wmma_kernel<K, NCOLS, TANH><<<blocks, 128, 0, s>>>(X, W, bias, Y, mtiles);
}

struct ConvP { const float *We, *Wl, *Wr, *att, *bias, *bl, *br; };

template<int K>
static void run_conv(const float* Xsrc, int Nsrc, const float* Xdst, int Ndst,
                     const ConvP& p,
                     const int* srcIdx, const int* dstIdx, const float* eattr,
                     float* out,
                     float* xsbuf, float* xdbuf, float* score, float* smax, float* denom,
                     hipStream_t s)
{
    launch_gemm<K, HC, 0>(Xsrc, p.Wl, p.bl, xsbuf, Nsrc, s);   // lin_l(source)
    launch_gemm<K, HC, 0>(Xdst, p.Wr, p.br, xdbuf, Ndst, s);   // lin_r(target)
    int n3 = Ndst * HEADS;
    fill_kernel<<<(n3 + 255) / 256, 256, 0, s>>>(smax, -INFINITY, n3);
    fill_kernel<<<(n3 + 255) / 256, 256, 0, s>>>(denom, 0.0f, n3);
    init_bias_kernel<<<(Ndst * HC + 255) / 256, 256, 0, s>>>(out, p.bias, Ndst);
    int eblocks = (N_EDGES * 32 + 255) / 256;   // wave per edge, 8 edges / block
    edge_score_kernel<<<eblocks, 256, 0, s>>>(srcIdx, dstIdx, eattr, p.We, p.att,
                                              xsbuf, xdbuf, score, smax, N_EDGES);
    edge_exp_kernel<<<(N_EDGES + 255) / 256, 256, 0, s>>>(dstIdx, score, smax, denom, N_EDGES);
    edge_scatter_kernel<<<eblocks, 256, 0, s>>>(srcIdx, dstIdx, xsbuf, score, denom, out, N_EDGES);
}

// ---------------------------------------------------------------------------
// Entry point
// ---------------------------------------------------------------------------
extern "C" void kernel_launch(void* const* d_in, const int* in_sizes, int n_in,
                              void* d_out, int out_size, void* d_ws, size_t ws_size,
                              hipStream_t stream)
{
    (void)in_sizes; (void)n_in; (void)out_size; (void)ws_size;

    // ---- inputs (setup_inputs() insertion order; params as jax-sorted pytree) ----
    const float* x_m     = (const float*)d_in[0];   // [50000,16]
    const float* x_j     = (const float*)d_in[1];   // [50000,16]
    const int*   ei      = (const int*)  d_in[2];   // [2,E]
    const float* ea_exec = (const float*)d_in[3];   // [E,5]
    const float* ea_rev  = (const float*)d_in[4];   // [E,5]
    const int* e_src = ei;             // machine ids
    const int* e_dst = ei + N_EDGES;   // job ids

    // params leaves, sorted-key pytree order:
    //  5:W3  6:b3
    //  per conv dict (keys sorted): We, Wl, Wr, att, bias, bl, br
    //  7..13  convs[0].exec   14..20 convs[0].rev
    //  21..27 convs[1].exec   28..34 convs[1].rev
    //  35:lin1_j_W 36:lin1_j_b 37:lin1_m_W 38:lin1_m_b
    const float* W3 = (const float*)d_in[5];
    const float* b3 = (const float*)d_in[6];
    ConvP cv[4];
    for (int i = 0; i < 4; ++i) {
        int b = 7 + 7 * i;
        cv[i].We   = (const float*)d_in[b + 0];
        cv[i].Wl   = (const float*)d_in[b + 1];
        cv[i].Wr   = (const float*)d_in[b + 2];
        cv[i].att  = (const float*)d_in[b + 3];
        cv[i].bias = (const float*)d_in[b + 4];
        cv[i].bl   = (const float*)d_in[b + 5];
        cv[i].br   = (const float*)d_in[b + 6];
    }
    const float* lin1_j_W = (const float*)d_in[35];
    const float* lin1_j_b = (const float*)d_in[36];
    const float* lin1_m_W = (const float*)d_in[37];
    const float* lin1_m_b = (const float*)d_in[38];

    float* out = (float*)d_out;        // [E] float32

    // ---- workspace carve (all float32) ----
    size_t off = 0;
    char* base = (char*)d_ws;
    auto carve = [&](size_t nfloats) -> float* {
        float* p = (float*)(base + off);
        off += nfloats * sizeof(float);
        return p;
    };
    float* hA_m  = carve((size_t)N_NODES * HC);   // ping buffers (layer in)
    float* hA_j  = carve((size_t)N_NODES * HC);
    float* hB_m  = carve((size_t)N_NODES * HC);   // pong buffers (layer out)
    float* hB_j  = carve((size_t)N_NODES * HC);
    float* xsbuf = carve((size_t)N_NODES * HC);   // per-conv lin_l output
    float* xdbuf = carve((size_t)N_NODES * HC);   // per-conv lin_r output
    float* score = carve((size_t)N_EDGES * HEADS);
    float* smax  = carve((size_t)N_NODES * HEADS);
    float* denom = carve((size_t)N_NODES * HEADS);
    float* pm    = carve((size_t)N_NODES);
    float* pj    = carve((size_t)N_NODES);

    // ---- 1) lin1 + tanh (WMMA, K=16 -> 8 cols, stored compact ld=8) ----
    launch_gemm<IN_FEAT, LIN1, 1>(x_m, lin1_m_W, lin1_m_b, hA_m, N_NODES, stream);
    launch_gemm<IN_FEAT, LIN1, 1>(x_j, lin1_j_W, lin1_j_b, hA_j, N_NODES, stream);

    // ---- 2) layer 0 (din=8): both convs read hA_*, write hB_* ----
    // new_j = gatv2(hm, hj, src=machines, dst=jobs, exec attrs)
    run_conv<LIN1>(hA_m, N_NODES, hA_j, N_NODES, cv[0],
                   e_src, e_dst, ea_exec, hB_j, xsbuf, xdbuf, score, smax, denom, stream);
    // new_m = gatv2(hj, hm, src=jobs, dst=machines, rev attrs)
    run_conv<LIN1>(hA_j, N_NODES, hA_m, N_NODES, cv[1],
                   e_dst, e_src, ea_rev, hB_m, xsbuf, xdbuf, score, smax, denom, stream);

    // ---- 3) layer 1 (din=96): read hB_*, write hA_* ----
    run_conv<HC>(hB_m, N_NODES, hB_j, N_NODES, cv[2],
                 e_src, e_dst, ea_exec, hA_j, xsbuf, xdbuf, score, smax, denom, stream);
    run_conv<HC>(hB_j, N_NODES, hB_m, N_NODES, cv[3],
                 e_dst, e_src, ea_rev, hA_m, xsbuf, xdbuf, score, smax, denom, stream);

    // ---- 4) actor head: per-node tanh+dot, then per-edge combine ----
    int nwb = (N_NODES * 32 + 255) / 256;
    node_dot_kernel<<<nwb, 256, 0, stream>>>(hA_m, W3 + 0,               pm, N_NODES);
    node_dot_kernel<<<nwb, 256, 0, stream>>>(hA_j, W3 + HC + EDGE_DIM,   pj, N_NODES);
    edge_out_kernel<<<(N_EDGES + 255) / 256, 256, 0, stream>>>(
        e_src, e_dst, ea_exec, W3, b3, pm, pj, out, N_EDGES);
}
